// HKLinear1D_29128468201623
// MI455X (gfx1250) — matrix operand order repaired and checked by reading, not compiled, not run on hardware
//
#include <hip/hip_runtime.h>
#include <hip/hip_bf16.h>

#define N_Q    4096
#define D_IN   512
#define D_OUT  16384
#define C_NUM  256
#define THRESH 0.01f
#define INV_T  10.0f   // 1 / TEMPERATURE

typedef __attribute__((ext_vector_type(16))) __bf16        v16bf;
typedef __attribute__((ext_vector_type(8)))  float         v8f;
typedef __attribute__((ext_vector_type(4)))  unsigned int  v4u;
typedef int i32x4 __attribute__((vector_size(16)));   // matches builtin param type

// ---------------------------------------------------------------------------
// f32 -> bf16 (RNE) pack helpers
// ---------------------------------------------------------------------------
__device__ __forceinline__ unsigned int pack2_bf16(float a, float b) {
#if __has_builtin(__builtin_amdgcn_cvt_pk_bf16_f32)
    return __builtin_bit_cast(unsigned int, __builtin_amdgcn_cvt_pk_bf16_f32(a, b));
#else
    unsigned int ua = __float_as_uint(a);
    unsigned int ub = __float_as_uint(b);
    ua += 0x7FFFu + ((ua >> 16) & 1u);
    ub += 0x7FFFu + ((ub >> 16) & 1u);
    return (ua >> 16) | (ub & 0xFFFF0000u);
#endif
}

// ---------------------------------------------------------------------------
// CDNA5 async global->LDS copy (ASYNCcnt-tracked) + wait
// ---------------------------------------------------------------------------
__device__ __forceinline__ void async_copy_b128(const unsigned short* gsrc,
                                                unsigned short* ldst) {
#if __has_builtin(__builtin_amdgcn_global_load_async_to_lds_b128)
    __builtin_amdgcn_global_load_async_to_lds_b128(
        (i32x4*)const_cast<unsigned short*>(gsrc),
        (__attribute__((address_space(3))) i32x4*)ldst, 0, 0);
#else
    asm volatile("global_load_async_to_lds_b128 %0, %1, off"
                 :: "v"((unsigned int)(__UINTPTR_TYPE__)
                        (__attribute__((address_space(3))) void*)ldst),
                    "v"(gsrc)
                 : "memory");
#endif
}

#if __has_builtin(__builtin_amdgcn_s_wait_asynccnt)
#define WAIT_ASYNC(n) __builtin_amdgcn_s_wait_asynccnt(n)
#else
#define WAIT_ASYNC(n) asm volatile("s_wait_asynccnt %0" :: "i"(n) : "memory")
#endif

// ---------------------------------------------------------------------------
// One-time f32 -> bf16 conversion pass (8 floats / thread)
// ---------------------------------------------------------------------------
__global__ void __launch_bounds__(256)
convert_bf16_kernel(const float* __restrict__ src, unsigned short* __restrict__ dst) {
    const size_t i = (size_t)blockIdx.x * blockDim.x + threadIdx.x;
    const float4* s = reinterpret_cast<const float4*>(src) + 2 * i;
    const float4 a = s[0];
    const float4 b = s[1];
    v4u o;
    o.x = pack2_bf16(a.x, a.y);
    o.y = pack2_bf16(a.z, a.w);
    o.z = pack2_bf16(b.x, b.y);
    o.w = pack2_bf16(b.z, b.w);
    *reinterpret_cast<v4u*>(dst + i * 8) = o;
}

// ---------------------------------------------------------------------------
// cmask must be zero at the start of every launch
// ---------------------------------------------------------------------------
__global__ void zero_cmask_kernel(int* __restrict__ cmask) {
    cmask[threadIdx.x] = 0;
}

// ---------------------------------------------------------------------------
// Routing: softmax(input @ centroids.T / T) > THRESH -> qmask[N], cmask[C]
// ---------------------------------------------------------------------------
__global__ void __launch_bounds__(256)
routing_kernel(const float* __restrict__ input,
               const float* __restrict__ centroids,
               int* __restrict__ qmask,
               int* __restrict__ cmask) {
    __shared__ float s_row[D_IN];
    __shared__ float s_red[C_NUM];
    __shared__ int   s_any;

    const int t = threadIdx.x;
    const int i = blockIdx.x;

    s_row[t]       = input[(size_t)i * D_IN + t];
    s_row[t + 256] = input[(size_t)i * D_IN + 256 + t];
    if (t == 0) s_any = 0;
    __syncthreads();

    const float4* cp = reinterpret_cast<const float4*>(centroids + (size_t)t * D_IN);
    const float4* xp = reinterpret_cast<const float4*>(s_row);
    float dot = 0.0f;
#pragma unroll 8
    for (int k = 0; k < D_IN / 4; ++k) {
        float4 c = cp[k];
        float4 x = xp[k];
        dot = fmaf(c.x, x.x, dot);
        dot = fmaf(c.y, x.y, dot);
        dot = fmaf(c.z, x.z, dot);
        dot = fmaf(c.w, x.w, dot);
    }
    dot *= INV_T;

    s_red[t] = dot;
    __syncthreads();
    for (int s = 128; s > 0; s >>= 1) {
        if (t < s) s_red[t] = fmaxf(s_red[t], s_red[t + s]);
        __syncthreads();
    }
    const float mx = s_red[0];
    __syncthreads();

    const float e = __expf(dot - mx);

    s_red[t] = e;
    __syncthreads();
    for (int s = 128; s > 0; s >>= 1) {
        if (t < s) s_red[t] += s_red[t + s];
        __syncthreads();
    }
    const float denom = s_red[0];

    const bool active = e > THRESH * denom;
    if (active) {
        cmask[t] = 1;   // idempotent racy store
        s_any = 1;
    }
    __syncthreads();
    if (t == 0) qmask[i] = s_any;
}

// ---------------------------------------------------------------------------
// Fused masked GEMM on pre-converted bf16:
//   out = (input @ weight.T + bias) * qmask[:,None] * rowmask[None,:]
// Block tile 128x256, 8 waves (2x4), wave tile 64x64 = 16 WMMA accumulators.
// Double-buffered LDS staged with global_load_async_to_lds_b128.
// ---------------------------------------------------------------------------
#define LDS_STRIDE 40        // ushorts/row: 32 data + 8 pad (80B, 16B-aligned)
#define A_BUF      (128 * LDS_STRIDE)
#define B_BUF      (256 * LDS_STRIDE)

__global__ void __launch_bounds__(256)
gemm_masked_kernel(const unsigned short* __restrict__ Abf,   // [N_Q, D_IN] bf16
                   const unsigned short* __restrict__ Wbf,   // [D_OUT, D_IN] bf16
                   const float* __restrict__ bias,
                   const int*  __restrict__ assignment,
                   const int*  __restrict__ qmask,
                   const int*  __restrict__ cmask,
                   float* __restrict__ out) {
    __shared__ unsigned short ldsA[2 * A_BUF];
    __shared__ unsigned short ldsB[2 * B_BUF];
    __shared__ float s_qf[128];
    __shared__ float s_rf[256];
    __shared__ float s_bias[256];

    const int t      = threadIdx.x;
    const int blockN = blockIdx.x;       // D_OUT / 256
    const int blockM = blockIdx.y;       // N_Q   / 128
    const int lane   = t & 31;
    const int wave   = t >> 5;
    const int waveM  = wave >> 2;        // 0..1  (64 rows)
    const int waveN  = wave & 3;         // 0..3  (64 cols)
    const int l15    = lane & 15;
    const int half   = lane >> 4;

    // per-block masks + bias
    if (t < 128) s_qf[t] = qmask[blockM * 128 + t] ? 1.0f : 0.0f;
    {
        const int gc = blockN * 256 + t;
        s_rf[t]   = cmask[assignment[gc]] ? 1.0f : 0.0f;
        s_bias[t] = bias[gc];
    }

    v8f acc[16];
#pragma unroll
    for (int i = 0; i < 16; ++i)
        acc[i] = (v8f){0.f, 0.f, 0.f, 0.f, 0.f, 0.f, 0.f, 0.f};

    const unsigned short* Ag = Abf + (size_t)(blockM * 128) * D_IN;
    const unsigned short* Bg = Wbf + (size_t)(blockN * 256) * D_IN;

    // async stage of one 32-wide K slice into buffer `buf` (6 x b128 / thread)
    auto stage = [&](int buf, int k0) {
#pragma unroll
        for (int p = 0; p < 2; ++p) {               // A: 128 rows x 4 chunks
            const int c = p * 256 + t;
            const int row = c >> 2, q = c & 3;
            async_copy_b128(Ag + (size_t)row * D_IN + k0 + q * 8,
                            &ldsA[buf * A_BUF + row * LDS_STRIDE + q * 8]);
        }
#pragma unroll
        for (int p = 0; p < 4; ++p) {               // B: 256 rows x 4 chunks
            const int c = p * 256 + t;
            const int row = c >> 2, q = c & 3;
            async_copy_b128(Bg + (size_t)row * D_IN + k0 + q * 8,
                            &ldsB[buf * B_BUF + row * LDS_STRIDE + q * 8]);
        }
    };

    stage(0, 0);

    for (int ks = 0; ks < D_IN / 32; ++ks) {
        const int buf = ks & 1;
        if (ks + 1 < D_IN / 32) {
            stage(buf ^ 1, (ks + 1) * 32);   // overlap next slice's DMA
            WAIT_ASYNC(6);                   // previous 6 copies complete
        } else {
            WAIT_ASYNC(0);
        }
        __syncthreads();                     // all waves' copies visible

        // fragments (ISA 7.12.2 16-bit A 16x32 / B 32x16 layouts)
        const unsigned short* bA = &ldsA[buf * A_BUF];
        const unsigned short* bB = &ldsB[buf * B_BUF];
        v16bf afrag[4];
#pragma unroll
        for (int am = 0; am < 4; ++am) {
            const int r = waveM * 64 + am * 16 + l15;
            const unsigned short* pa = bA + r * LDS_STRIDE + half * 8;
            v4u x0 = *reinterpret_cast<const v4u*>(pa);
            v4u x1 = *reinterpret_cast<const v4u*>(pa + 16);
            afrag[am] = __builtin_bit_cast(
                v16bf, __builtin_shufflevector(x0, x1, 0, 1, 2, 3, 4, 5, 6, 7));
        }
        v16bf bfrag[4];
#pragma unroll
        for (int bn = 0; bn < 4; ++bn) {
            const int c = waveN * 64 + bn * 16 + l15;
            const unsigned short* pb = bB + c * LDS_STRIDE + half * 8;
            v4u y0 = *reinterpret_cast<const v4u*>(pb);
            v4u y1 = *reinterpret_cast<const v4u*>(pb + 16);
            bfrag[bn] = __builtin_bit_cast(
                v16bf, __builtin_shufflevector(y0, y1, 0, 1, 2, 3, 4, 5, 6, 7));
        }

        // 16 x v_wmma_f32_16x16x32_bf16
#pragma unroll
        for (int am = 0; am < 4; ++am) {
#pragma unroll
            for (int bn = 0; bn < 4; ++bn) {
                acc[am * 4 + bn] = __builtin_amdgcn_wmma_f32_16x16x32_bf16(
                    false, afrag[am], false, bfrag[bn],
                    (short)0, acc[am * 4 + bn], false, false);
            }
        }
        __syncthreads();   // reads done before next overwrite of this buffer
    }

    // fused epilogue: (acc + bias) * qmask * rowmask
#pragma unroll
    for (int am = 0; am < 4; ++am) {
#pragma unroll
        for (int bn = 0; bn < 4; ++bn) {
            const int    colOff = waveN * 64 + bn * 16 + l15;
            const float  rfv    = s_rf[colOff];
            const float  bv     = s_bias[colOff];
            const size_t gcol   = (size_t)blockN * 256 + colOff;
            const v8f    a      = acc[am * 4 + bn];
#pragma unroll
            for (int r = 0; r < 8; ++r) {
                const int rowOff = waveM * 64 + am * 16 + half * 8 + r;
                const float v    = (a[r] + bv) * s_qf[rowOff] * rfv;
                out[(size_t)(blockM * 128 + rowOff) * D_OUT + gcol] = v;
            }
        }
    }
}

// ---------------------------------------------------------------------------
extern "C" void kernel_launch(void* const* d_in, const int* in_sizes, int n_in,
                              void* d_out, int out_size, void* d_ws, size_t ws_size,
                              hipStream_t stream) {
    const float* input      = (const float*)d_in[0];   // [N, D_IN]
    const float* weight     = (const float*)d_in[1];   // [D_OUT, D_IN]
    const float* bias       = (const float*)d_in[2];   // [D_OUT]
    const float* centroids  = (const float*)d_in[3];   // [C, D_IN]
    const int*   assignment = (const int*)  d_in[4];   // [D_OUT]
    float*       out        = (float*)d_out;           // [N, D_OUT]

    // workspace: [qmask 16KB][cmask 1KB][pad][A_bf16 4MB][W_bf16 16MB]
    int* qmask = (int*)d_ws;
    int* cmask = qmask + N_Q;
    unsigned short* Abf = (unsigned short*)((char*)d_ws + 32768);
    unsigned short* Wbf = Abf + (size_t)N_Q * D_IN;

    // one-time f32 -> bf16 conversion (8 floats / thread)
    convert_bf16_kernel<<<(N_Q * D_IN) / (8 * 256), 256, 0, stream>>>(input, Abf);
    convert_bf16_kernel<<<((size_t)D_OUT * D_IN) / (8 * 256), 256, 0, stream>>>(weight, Wbf);

    zero_cmask_kernel<<<1, C_NUM, 0, stream>>>(cmask);
    routing_kernel<<<N_Q, C_NUM, 0, stream>>>(input, centroids, qmask, cmask);

    dim3 grid(D_OUT / 256, N_Q / 128);   // (64, 32)
    gemm_masked_kernel<<<grid, 256, 0, stream>>>(Abf, Wbf, bias,
                                                 assignment, qmask, cmask, out);
}
